// SpatioTemporalMask_82686710383310
// MI455X (gfx1250) — matrix operand order repaired and checked by reading, not compiled, not run on hardware
//
#include <hip/hip_runtime.h>

// Problem constants (from reference)
#define T_ 6
#define N_ 50000
#define E_ 100000
#define D_ 128
#define H_ 128

typedef __attribute__((ext_vector_type(16))) _Float16 v16h;
typedef __attribute__((ext_vector_type(8)))  _Float16 v8h;
typedef __attribute__((ext_vector_type(8)))  float    v8f;

// ---------------- LDS layout (bytes) ----------------
// W1 swizzled to B-fragment order: [nt(8)][k(8)][lane(32)][16 halves]
constexpr int W1_HALVES   = 8 * 8 * 32 * 16;   // 32768 -> 64 KB
// W2 swizzled: [nt(8)][k(4)][lane(32)][16 halves]
constexpr int W2_HALVES   = 8 * 4 * 32 * 16;   // 16384 -> 32 KB
constexpr int A_STRIDE_H  = 264;               // 16x256 tile, padded row (breaks bank conflicts)
constexpr int A_TILE_B    = 16 * A_STRIDE_H * 2; // 8448 B per wave
constexpr int H1_STRIDE_H = 136;               // padded 16x128 h1 tile (reuses A region)

constexpr int OFF_W1 = 0;
constexpr int OFF_W2 = OFF_W1 + W1_HALVES * 2;   // 65536
constexpr int OFF_A  = OFF_W2 + W2_HALVES * 2;   // 98304
constexpr int OFF_F  = OFF_A + 8 * A_TILE_B;     // 165888 (float tables)
// float tables: W3[128] @ +0, b1[128] @ +512, b2[128] @ +1024, W1_time_row[128] @ +1536
constexpr int SMEM_BYTES = OFF_F + 2048;         // 167936 B (< 320 KB WGP LDS)

__device__ __forceinline__ v16h cat8(v8h lo, v8h hi) {
  return __builtin_shufflevector(lo, hi, 0,1,2,3,4,5,6,7,8,9,10,11,12,13,14,15);
}

__global__ __launch_bounds__(256)
void edge_mlp_kernel(const float* __restrict__ nf, const int* __restrict__ ei,
                     const float* __restrict__ ts, const float* __restrict__ noise,
                     const float* __restrict__ W1, const float* __restrict__ b1,
                     const float* __restrict__ W2, const float* __restrict__ b2,
                     const float* __restrict__ W3, const float* __restrict__ b3,
                     float* __restrict__ m_out) {
  extern __shared__ char smem[];
  const int tid  = threadIdx.x;
  const int lane = tid & 31;
  const int wave = tid >> 5;
  const int t    = blockIdx.y;

  // ---- cooperative weight pre-swizzle into B-fragment order (one-time) ----
  _Float16* sW1 = (_Float16*)(smem + OFF_W1);
  for (int idx = tid; idx < W1_HALVES; idx += 256) {
    int i  = idx & 15;
    int ln = (idx >> 4) & 31;
    int k  = (idx >> 9) & 7;
    int nt = idx >> 12;
    int K  = 32 * k + ((i >> 3) << 4) + ((ln >= 16) ? 8 : 0) + (i & 7);     // input-feature index
    int Nc = 16 * nt + (ln & 15);                    // hidden column
    sW1[idx] = (_Float16)W1[K * H_ + Nc];
  }
  _Float16* sW2 = (_Float16*)(smem + OFF_W2);
  for (int idx = tid; idx < W2_HALVES; idx += 256) {
    int i  = idx & 15;
    int ln = (idx >> 4) & 31;
    int k  = (idx >> 9) & 3;
    int nt = idx >> 11;
    int K  = 32 * k + ((i >> 3) << 4) + ((ln >= 16) ? 8 : 0) + (i & 7);
    int Nc = 16 * nt + (ln & 15);
    sW2[idx] = (_Float16)W2[K * H_ + Nc];
  }
  float* sW3f  = (float*)(smem + OFF_F);
  float* sB1f  = (float*)(smem + OFF_F + 512);
  float* sB2f  = (float*)(smem + OFF_F + 1024);
  float* sW1tf = (float*)(smem + OFF_F + 1536);
  if (tid < 128) {
    sW3f[tid]  = W3[tid];                // (128,1)
    sB1f[tid]  = b1[tid];
    sB2f[tid]  = b2[tid];
    sW1tf[tid] = W1[256 * H_ + tid];     // time-feature row of W1 (folded into bias)
  }
  __syncthreads();

  const int tile = blockIdx.x * 8 + wave;          // 16-edge tile index within this t
  if (tile >= E_ / 16) return;                     // E divisible by 16; tail waves idle
  const int e0 = tile * 16;

  // ---- gather: 16 edges x [src|dst] (256 f32) -> f16 A-tile in LDS ----
  {
    const int r  = lane & 15;        // edge row within tile
    const int hf = lane >> 4;        // 0 = src half, 1 = dst half
    const int nidx = ei[((size_t)t * 2 + hf) * E_ + (e0 + r)];
    const float* rowp = nf + ((size_t)t * N_ + nidx) * D_;
    _Float16* ap = (_Float16*)(smem + OFF_A + wave * A_TILE_B) + r * A_STRIDE_H + hf * 128;
#pragma unroll
    for (int c = 0; c < 128; c += 8) {
      float4 f0 = *(const float4*)(rowp + c);
      float4 f1 = *(const float4*)(rowp + c + 4);
      v8f fv = {f0.x, f0.y, f0.z, f0.w, f1.x, f1.y, f1.z, f1.w};
      *(v8h*)(ap + c) = __builtin_convertvector(fv, v8h);
    }
  }

  // ---- load all layer-1 A fragments into registers (frees LDS region for h1) ----
  const char* aBase = smem + OFF_A + wave * A_TILE_B;
  const int arow = lane & 15;
  const int qoff = (lane >> 4) << 3;   // hi half-wave holds K+8 / K+24 slices
  v16h afrag[8];
#pragma unroll
  for (int k = 0; k < 8; ++k) {
    const v8h* p0 = (const v8h*)(aBase + (arow * A_STRIDE_H + 32 * k + qoff) * 2);
    const v8h* p1 = (const v8h*)(aBase + (arow * A_STRIDE_H + 32 * k + 16 + qoff) * 2);
    afrag[k] = cat8(*p0, *p1);
  }

  // ---- layer 1: [16x256] @ [256x128] via 64 WMMAs, relu, restage h1 in LDS ----
  const float tval = ts[t];
  const int nc = lane & 15;
  _Float16* sH = (_Float16*)(smem + OFF_A + wave * A_TILE_B);   // reuse A region
#pragma unroll
  for (int nt = 0; nt < 8; ++nt) {
    const int Nc = 16 * nt + nc;
    const float bias = sB1f[Nc] + tval * sW1tf[Nc];   // fold time feature into bias
    v8f c = {bias, bias, bias, bias, bias, bias, bias, bias};
#pragma unroll
    for (int k = 0; k < 8; ++k) {
      const v8h* bp = (const v8h*)(smem + OFF_W1 + (((nt * 8 + k) * 32 + lane) << 5));
      v16h bfrag = cat8(bp[0], bp[1]);
      c = __builtin_amdgcn_wmma_f32_16x16x32_f16(false, afrag[k], false, bfrag,
                                                 (short)0, c, false, false);
    }
    const int mbase = (lane >> 4) << 3;
#pragma unroll
    for (int j = 0; j < 8; ++j) {
      sH[(mbase + j) * H1_STRIDE_H + Nc] = (_Float16)fmaxf(c[j], 0.0f);
    }
  }

  // ---- layer-2 A fragments from h1 ----
  v16h a2[4];
#pragma unroll
  for (int k = 0; k < 4; ++k) {
    const v8h* p0 = (const v8h*)((const char*)sH + (arow * H1_STRIDE_H + 32 * k + qoff) * 2);
    const v8h* p1 = (const v8h*)((const char*)sH + (arow * H1_STRIDE_H + 32 * k + 16 + qoff) * 2);
    a2[k] = cat8(*p0, *p1);
  }

  // ---- layer 2 (32 WMMAs) fused with layer-3 dot against W3 ----
  float pacc[8] = {0.f, 0.f, 0.f, 0.f, 0.f, 0.f, 0.f, 0.f};
#pragma unroll
  for (int nt = 0; nt < 8; ++nt) {
    const int Nc = 16 * nt + nc;
    const float bias2 = sB2f[Nc];
    v8f c = {bias2, bias2, bias2, bias2, bias2, bias2, bias2, bias2};
#pragma unroll
    for (int k = 0; k < 4; ++k) {
      const v8h* bp = (const v8h*)(smem + OFF_W2 + (((nt * 4 + k) * 32 + lane) << 5));
      v16h bfrag = cat8(bp[0], bp[1]);
      c = __builtin_amdgcn_wmma_f32_16x16x32_f16(false, a2[k], false, bfrag,
                                                 (short)0, c, false, false);
    }
    const float w3v = sW3f[Nc];
#pragma unroll
    for (int j = 0; j < 8; ++j) pacc[j] += fmaxf(c[j], 0.0f) * w3v;
  }

  // ---- reduce over the 16 N-lanes of each C half (wave32 butterfly) ----
#pragma unroll
  for (int j = 0; j < 8; ++j) {
    float v = pacc[j];
    v += __shfl_xor(v, 1);
    v += __shfl_xor(v, 2);
    v += __shfl_xor(v, 4);
    v += __shfl_xor(v, 8);
    pacc[j] = v;   // every lane in the 16-lane half now holds raw[M]
  }

  // ---- mask math + pre-scan write (lane j -> M=j, lane 16+j -> M=j+8) ----
  const float b3v = b3[0];
  const int jm = lane & 15;
#pragma unroll
  for (int j = 0; j < 8; ++j) {
    if (jm == j) {
      const int M = j + ((lane >= 16) ? 8 : 0);
      const int e = e0 + M;
      const float raw = pacc[j] + b3v;
      const float sig = 1.0f / (1.0f + __expf(-raw * 5.0f));   // raw / 0.2
      float mv = sig * 0.8f + 0.2f * tval;
      mv = fminf(fmaxf(mv, 0.05f), 0.95f);
      mv += noise[(size_t)t * E_ + e] * 0.1f;
      mv = fminf(fmaxf(mv, 0.05f), 0.95f);
      m_out[(size_t)t * E_ + e] = mv;
    }
  }
}

// In-place temporal EMA scan over T=6 steps; each thread owns one edge column.
__global__ __launch_bounds__(256)
void scan_kernel(float* __restrict__ m) {
  const int e = blockIdx.x * 256 + threadIdx.x;
  if (e >= E_) return;
  float prev = m[e];                       // out[0] = m[0]
#pragma unroll
  for (int t = 1; t < T_; ++t) {
    const float cur = m[(size_t)t * E_ + e];
    prev = fminf(fmaxf(cur * 0.8f + prev * 0.2f, 0.05f), 0.95f);
    m[(size_t)t * E_ + e] = prev;
  }
}

extern "C" void kernel_launch(void* const* d_in, const int* in_sizes, int n_in,
                              void* d_out, int out_size, void* d_ws, size_t ws_size,
                              hipStream_t stream) {
  (void)in_sizes; (void)n_in; (void)out_size; (void)d_ws; (void)ws_size;
  const float* nf    = (const float*)d_in[0];
  const int*   ei    = (const int*)d_in[1];
  const float* ts    = (const float*)d_in[2];
  const float* noise = (const float*)d_in[3];
  const float* W1    = (const float*)d_in[4];
  const float* b1    = (const float*)d_in[5];
  const float* W2    = (const float*)d_in[6];
  const float* b2    = (const float*)d_in[7];
  const float* W3    = (const float*)d_in[8];
  const float* b3    = (const float*)d_in[9];
  float* out = (float*)d_out;

  dim3 grid((E_ / 16 + 7) / 8, T_, 1);   // (782, 6): 8 waves/block, 16 edges/wave
  edge_mlp_kernel<<<grid, 256, SMEM_BYTES, stream>>>(nf, ei, ts, noise,
                                                     W1, b1, W2, b2, W3, b3, out);
  scan_kernel<<<(E_ + 255) / 256, 256, 0, stream>>>(out);
}
